// NNConvPerformanceModel_90623809946057
// MI455X (gfx1250) — matrix operand order repaired and checked by reading, not compiled, not run on hardware
//
#include <hip/hip_runtime.h>
#include <hip/hip_bf16.h>

typedef __attribute__((ext_vector_type(16))) __bf16 v16bf;
typedef __attribute__((ext_vector_type(8)))  __bf16 v8bf;
typedef __attribute__((ext_vector_type(8)))  float  v8f;

namespace {
constexpr int NODES  = 16384;
constexpr int EDGES  = 32768;
constexpr int GRAPHS = 8;
constexpr int MT     = 128;   // edges per workgroup (8 WMMA M-subtiles)
constexpr int TPB    = 192;   // 6 wave32 -> N=96 covered (6 x 16 cols)
constexpr int KSPLIT = 4;     // split K'' across gridDim.y (atomic scatter makes it legal)
}

// ---------------------------------------------------------------------------
// Permute + convert wb (f32, d x d) into bf16 "B''" fragment-major layout.
// B''[c, o] = wb[k, i*96+o] with c = k*CI+i (k-major, CI=96) or c = i*D+k
// (i-major, CI=6). Storage: per (32-row x 16-col) WMMA B fragment, 512 bf16
// contiguous, lane L owns elements [L*16 .. L*16+15]:
//   col n = L&15, K-in-frag = (L>=16 ? 8 : 0) + (e < 8 ? e : e + 8).
// ---------------------------------------------------------------------------
template<int CI, int D, bool IMAJOR>
__global__ void k_convert_wb(const float* __restrict__ wb, __bf16* __restrict__ out)
{
  long long tid = (long long)blockIdx.x * 256 + threadIdx.x;   // D*D elements
  long long blk = tid >> 9;
  int r  = (int)(tid & 511);
  int L  = r >> 4, e = r & 15;
  long long kb = blk / 6;            // 6 col-blocks cover N=96
  int jb = (int)(blk % 6);
  int K  = ((L & 16) >> 1) + (e < 8 ? e : e + 8);
  int n  = L & 15;
  long long c = kb * 32 + K;
  int o = jb * 16 + n;
  long long k, i;
  if (IMAJOR) { i = c / D;  k = c % D;  }
  else        { k = c / CI; i = c % CI; }
  out[tid] = (__bf16)wb[k * (long long)D + i * 96 + o];
}

__global__ void k_xcast(const float* __restrict__ in, __bf16* __restrict__ out, long long n)
{
  long long t = (long long)blockIdx.x * 256 + threadIdx.x;
  if (t < n) out[t] = (__bf16)in[t];
}

// ---------------------------------------------------------------------------
// Fused NNConv message GEMM:  msg = A'' @ B''  (M=E, K''=CI*D, N=96)
//   A''[e, c] = x[src[e], i(c)] * relu(ea[e] . wa[:,k(c)] + ba[k(c)])
// generated in-register; accumulated in f32 via v_wmma_f32_16x16x32_bf16 and
// atomically scattered into sacc[dst[e]*96+o] (bb bias folded by kslice 0).
// x tile is gathered into LDS with async global->LDS DMA (bf16 mirror);
// A''-fragment scaling pinned to v_pk_mul_bf16 / v_cvt_pk_bf16_f32.
// ---------------------------------------------------------------------------
template<int CI, int D>
__global__ __launch_bounds__(TPB) void k_edge_gemm(
    const float* __restrict__ ea,  const float* __restrict__ wa,
    const float* __restrict__ ba,  const __bf16* __restrict__ xinb,
    const int*   __restrict__ src, const int*   __restrict__ dst,
    const __bf16* __restrict__ wbp, const float* __restrict__ bb,
    float* __restrict__ sacc)
{
  constexpr bool IMAJOR = (CI < 32);
  constexpr int  CIP = (CI == 96) ? 104 : 8;   // padded LDS row (bank-conflict-free)
  constexpr long long KK = (long long)CI * D;
  constexpr int NCH = (int)(KK / 32);
  constexpr int PER = NCH / KSPLIT;
  static_assert(D % 32 == 0 && NCH % KSPLIT == 0, "tiling");

  __shared__ alignas(16) __bf16 xls[MT * CIP];  // gathered node features (bf16)
  __shared__ alignas(16) float  eals[MT * 3];   // edge attrs
  __shared__ alignas(16) float  wals[IMAJOR ? 3 * D : 4];
  __shared__ alignas(16) float  bals[IMAJOR ? D     : 4];

  const int tid = threadIdx.x;
  const long long ebase = (long long)blockIdx.x * MT;

  // ---- async DMA staging (global -> LDS, no VGPR round-trip) ----
  if constexpr (!IMAJOR) {
    for (int idx = tid; idx < MT * 12; idx += TPB) {        // 12 x b128 per row
      int m = idx / 12, c = idx % 12;
      int sn = src[ebase + m];
      const __bf16* g = xinb + (long long)sn * 96 + c * 8;
      unsigned l = (unsigned)(uintptr_t)&xls[m * CIP + c * 8];
      asm volatile("global_load_async_to_lds_b128 %0, %1, off" :: "v"(l), "v"(g) : "memory");
    }
  } else {
    for (int idx = tid; idx < MT * 3; idx += TPB) {         // 3 x b32 per row (6 bf16)
      int m = idx / 3, c = idx % 3;
      int sn = src[ebase + m];
      const __bf16* g = xinb + (long long)sn * 6 + c * 2;
      unsigned l = (unsigned)(uintptr_t)&xls[m * CIP + c * 2];
      asm volatile("global_load_async_to_lds_b32 %0, %1, off" :: "v"(l), "v"(g) : "memory");
    }
  }
  for (int idx = tid; idx < MT * 3; idx += TPB) {
    const float* g = ea + ebase * 3 + idx;
    unsigned l = (unsigned)(uintptr_t)&eals[idx];
    asm volatile("global_load_async_to_lds_b32 %0, %1, off" :: "v"(l), "v"(g) : "memory");
  }
  if constexpr (IMAJOR) {
    for (int idx = tid; idx < (3 * D) / 4; idx += TPB) {
      const float* g = wa + idx * 4;
      unsigned l = (unsigned)(uintptr_t)&wals[idx * 4];
      asm volatile("global_load_async_to_lds_b128 %0, %1, off" :: "v"(l), "v"(g) : "memory");
    }
    for (int idx = tid; idx < D / 4; idx += TPB) {
      const float* g = ba + idx * 4;
      unsigned l = (unsigned)(uintptr_t)&bals[idx * 4];
      asm volatile("global_load_async_to_lds_b128 %0, %1, off" :: "v"(l), "v"(g) : "memory");
    }
  }
  asm volatile("s_wait_asynccnt 0x0" ::: "memory");
  __syncthreads();

  const int w     = tid >> 5;      // wave -> output cols [w*16, w*16+16)
  const int lane  = tid & 31;
  const int lhalf = lane >> 4;
  const int lrow  = lane & 15;

  v8f acc[MT / 16];
  for (int s = 0; s < MT / 16; ++s)
    for (int r = 0; r < 8; ++r) acc[s][r] = 0.f;

  const int cb0 = blockIdx.y * PER;

  if constexpr (!IMAJOR) {
    // ---- fast path (CI=96): chunk has fixed k; hoist per-row state ----
    constexpr int ICN = CI / 32;             // 3
    const int nk = PER / ICN;
    const int kbase = cb0 / ICN;
    float ea0[MT / 16], ea1[MT / 16], ea2[MT / 16];
    const __bf16* xbase[MT / 16];
    #pragma unroll
    for (int s = 0; s < MT / 16; ++s) {
      int m = s * 16 + lrow;
      ea0[s] = eals[m * 3]; ea1[s] = eals[m * 3 + 1]; ea2[s] = eals[m * 3 + 2];
      xbase[s] = &xls[m * CIP + lhalf * 8];
    }
    const __bf16* bp = wbp + ((long long)cb0 * 6 + w) * 512 + lane * 16;
    v16bf bv = *(const v16bf*)bp;            // double-buffered B'' fragment
    for (int kk = 0; kk < nk; ++kk) {
      const int k = kbase + kk;
      const float wa0 = wa[k], wa1 = wa[D + k], wa2 = wa[2 * D + k], bak = ba[k];
      unsigned hp[MT / 16];                  // packed {h,h} bf16 broadcast per row
      #pragma unroll
      for (int s = 0; s < MT / 16; ++s) {
        float hf = fmaxf(ea0[s] * wa0 + ea1[s] * wa1 + ea2[s] * wa2 + bak, 0.f);
        asm("v_cvt_pk_bf16_f32 %0, %1, %1" : "=v"(hp[s]) : "v"(hf));
      }
      #pragma unroll
      for (int ic = 0; ic < ICN; ++ic) {
        v16bf bcur = bv;
        bp += 6 * 512;
        if (ic != ICN - 1 || kk != nk - 1) bv = *(const v16bf*)bp;
        __builtin_prefetch(bp + 4 * 6 * 512, 0, 1);   // global_prefetch_b8, 4 chunks ahead
        const int off = ic * 32;
        #pragma unroll
        for (int s = 0; s < MT / 16; ++s) {
          uint4 lo = *(const uint4*)(xbase[s] + off);        // ds_load_b128
          uint4 hi = *(const uint4*)(xbase[s] + off + 16);   // ds_load_b128
          union { v16bf v; unsigned u[8]; } av;
          asm("v_pk_mul_bf16 %0, %1, %2" : "=v"(av.u[0]) : "v"(lo.x), "v"(hp[s]));
          asm("v_pk_mul_bf16 %0, %1, %2" : "=v"(av.u[1]) : "v"(lo.y), "v"(hp[s]));
          asm("v_pk_mul_bf16 %0, %1, %2" : "=v"(av.u[2]) : "v"(lo.z), "v"(hp[s]));
          asm("v_pk_mul_bf16 %0, %1, %2" : "=v"(av.u[3]) : "v"(lo.w), "v"(hp[s]));
          asm("v_pk_mul_bf16 %0, %1, %2" : "=v"(av.u[4]) : "v"(hi.x), "v"(hp[s]));
          asm("v_pk_mul_bf16 %0, %1, %2" : "=v"(av.u[5]) : "v"(hi.y), "v"(hp[s]));
          asm("v_pk_mul_bf16 %0, %1, %2" : "=v"(av.u[6]) : "v"(hi.z), "v"(hp[s]));
          asm("v_pk_mul_bf16 %0, %1, %2" : "=v"(av.u[7]) : "v"(hi.w), "v"(hp[s]));
          acc[s] = __builtin_amdgcn_wmma_f32_16x16x32_bf16(
              false, av.v, false, bcur, (short)0, acc[s], false, false);
        }
      }
    }
  } else {
    // ---- generic path (CI=6, i-major): chunk has fixed i ----
    for (int cc = 0; cc < PER; ++cc) {
      const int cb = cb0 + cc;
      const __bf16* bp = wbp + ((long long)cb * 6 + w) * 512 + lane * 16;
      v16bf bv = *(const v16bf*)bp;
      __builtin_prefetch(bp + 6 * 512, 0, 1);
      const int i  = cb / (D / 32);
      const int k0 = (cb % (D / 32)) * 32;
      #pragma unroll
      for (int s = 0; s < MT / 16; ++s) {
        const int m = s * 16 + lrow;
        const float xi = (float)xls[m * CIP + i];
        const float e0 = eals[m * 3], e1 = eals[m * 3 + 1], e2 = eals[m * 3 + 2];
        v16bf av;
        #pragma unroll
        for (int e = 0; e < 16; ++e) {
          const int k = k0 + (e < 8 ? e : e + 8) + lhalf * 8;
          float h = e0 * wals[k] + e1 * wals[D + k] + e2 * wals[2 * D + k] + bals[k];
          av[e] = (__bf16)(xi * fmaxf(h, 0.f));
        }
        acc[s] = __builtin_amdgcn_wmma_f32_16x16x32_bf16(
            false, av, false, bv, (short)0, acc[s], false, false);
      }
    }
  }

  // Epilogue: C/D layout reg r -> row = r + 8*lhalf, col = lane&15.
  const int o = w * 16 + lrow;
  for (int s = 0; s < MT / 16; ++s) {
    #pragma unroll
    for (int r = 0; r < 8; ++r) {
      const int m = s * 16 + r + lhalf * 8;
      float v = acc[s][r];
      if (blockIdx.y == 0) {                     // fold bb once: sum_i x_e[i]*bb[i*96+o]
        float bsum = 0.f;
        for (int i = 0; i < CI; ++i)
          bsum += (float)xls[m * CIP + i] * bb[i * 96 + o];
        v += bsum;
      }
      const int d = dst[ebase + m];
      atomicAdd(&sacc[(long long)d * 96 + o], v);
    }
  }
}

__global__ void k_deg(const int* __restrict__ dst, float* __restrict__ deg)
{
  int t = blockIdx.x * 256 + threadIdx.x;
  if (t < EDGES) atomicAdd(&deg[dst[t]], 1.f);
}

template<int CI>
__global__ void k_node_update(const float* __restrict__ xin, const float* __restrict__ root,
                              const float* __restrict__ bias, const float* __restrict__ sacc,
                              const float* __restrict__ deg,  float* __restrict__ xoutf,
                              __bf16* __restrict__ xoutb)
{
  long long t = (long long)blockIdx.x * 256 + threadIdx.x;   // N*96 threads
  if (t >= (long long)NODES * 96) return;
  int n = (int)(t / 96), o = (int)(t % 96);
  float a = 0.f;
  for (int i = 0; i < CI; ++i) a += xin[(long long)n * CI + i] * root[i * 96 + o];
  float dg = deg[n]; dg = dg > 1.f ? dg : 1.f;
  float v = fmaxf(a + sacc[t] / dg + bias[o], 0.f);
  xoutf[t] = v;
  xoutb[t] = (__bf16)v;     // bf16 mirror consumed by next layer's WMMA gather
}

__global__ void k_pool(const float* __restrict__ h, const int* __restrict__ batch,
                       float* __restrict__ psum, float* __restrict__ cnt)
{
  long long t = (long long)blockIdx.x * 256 + threadIdx.x;
  if (t >= (long long)NODES * 96) return;
  int n = (int)(t / 96), o = (int)(t % 96);
  int g = batch[n];
  atomicAdd(&psum[g * 96 + o], h[t]);
  if (o == 0) atomicAdd(&cnt[g], 1.f);
}

__global__ void k_readout(const float* __restrict__ psum, const float* __restrict__ cnt,
                          const float* __restrict__ wm1, const float* __restrict__ bm1,
                          const float* __restrict__ wm2, const float* __restrict__ bm2,
                          float* __restrict__ out)
{
  __shared__ float pm[GRAPHS * 96], r1[GRAPHS * 96];
  int t = threadIdx.x;                 // GRAPHS*96 = 768 threads
  int g = t / 96, j = t % 96;
  float c = cnt[g]; c = c > 1.f ? c : 1.f;
  pm[t] = psum[t] / c;
  __syncthreads();
  float a = bm1[j];
  for (int i = 0; i < 96; ++i) a += pm[g * 96 + i] * wm1[i * 96 + j];
  r1[t] = fmaxf(a, 0.f);
  __syncthreads();
  if (j < 4) {
    float a2 = bm2[j];
    for (int i = 0; i < 96; ++i) a2 += r1[g * 96 + i] * wm2[i * 4 + j];
    out[g * 4 + j] = a2;
  }
}

extern "C" void kernel_launch(void* const* d_in, const int* in_sizes, int n_in,
                              void* d_out, int out_size, void* d_ws, size_t ws_size,
                              hipStream_t stream)
{
  (void)in_sizes; (void)n_in; (void)out_size; (void)ws_size;
  const float* x0    = (const float*)d_in[0];
  const int*   ei    = (const int*)  d_in[1];
  const float* eatt  = (const float*)d_in[2];
  const int*   batch = (const int*)  d_in[3];
  const float* w1a = (const float*)d_in[4],  *b1a  = (const float*)d_in[5];
  const float* w1b = (const float*)d_in[6],  *b1b  = (const float*)d_in[7];
  const float* rt1 = (const float*)d_in[8],  *bs1  = (const float*)d_in[9];
  const float* w2a = (const float*)d_in[10], *b2a  = (const float*)d_in[11];
  const float* w2b = (const float*)d_in[12], *b2b  = (const float*)d_in[13];
  const float* rt2 = (const float*)d_in[14], *bs2  = (const float*)d_in[15];
  const float* w3a = (const float*)d_in[16], *b3a  = (const float*)d_in[17];
  const float* w3b = (const float*)d_in[18], *b3b  = (const float*)d_in[19];
  const float* rt3 = (const float*)d_in[20], *bs3  = (const float*)d_in[21];
  const float* wm1 = (const float*)d_in[22], *bm1  = (const float*)d_in[23];
  const float* wm2 = (const float*)d_in[24], *bm2  = (const float*)d_in[25];
  const int* src = ei;
  const int* dst = ei + EDGES;

  // workspace carve-out (~370 MB): wbp2/wbp3 (170 MB each, L2-resident as bf16)
  char* ws = (char*)d_ws;
  size_t off = 0;
  auto take = [&](size_t bytes) {
    char* p = ws + off; off = (off + bytes + 255) & ~(size_t)255; return p;
  };
  __bf16* wbp1 = (__bf16*)take((size_t)576 * 576 * 2);
  __bf16* wbp2 = (__bf16*)take((size_t)9216 * 9216 * 2);
  __bf16* wbp3 = (__bf16*)take((size_t)9216 * 9216 * 2);
  float*  xA   = (float*)take((size_t)NODES * 96 * 4);
  float*  xB   = (float*)take((size_t)NODES * 96 * 4);
  float*  sb   = (float*)take((size_t)NODES * 96 * 4);
  float*  deg  = (float*)take((size_t)NODES * 4);
  float*  psum = (float*)take((size_t)GRAPHS * 96 * 4);
  float*  cnt  = (float*)take((size_t)GRAPHS * 4);
  __bf16* x0b  = (__bf16*)take((size_t)NODES * 6 * 2);
  __bf16* xAb  = (__bf16*)take((size_t)NODES * 96 * 2);
  __bf16* xBb  = (__bf16*)take((size_t)NODES * 96 * 2);

  // one-time (per call) weight permute/convert + layer-1 input cast
  k_convert_wb<6, 576, true>  <<<(576 * 576) / 256,            256, 0, stream>>>(w1b, wbp1);
  k_convert_wb<96, 9216, false><<<(int)((9216LL * 9216) / 256), 256, 0, stream>>>(w2b, wbp2);
  k_convert_wb<96, 9216, false><<<(int)((9216LL * 9216) / 256), 256, 0, stream>>>(w3b, wbp3);
  k_xcast<<<(NODES * 6 + 255) / 256, 256, 0, stream>>>(x0, x0b, (long long)NODES * 6);

  (void)hipMemsetAsync(deg, 0, (size_t)NODES * 4, stream);
  k_deg<<<EDGES / 256, 256, 0, stream>>>(dst, deg);

  const dim3 gg(EDGES / MT, KSPLIT);
  const int nu_blocks = (NODES * 96) / 256;

  // layer 1 (ci=6)
  (void)hipMemsetAsync(sb, 0, (size_t)NODES * 96 * 4, stream);
  k_edge_gemm<6, 576><<<gg, TPB, 0, stream>>>(eatt, w1a, b1a, x0b, src, dst, wbp1, b1b, sb);
  k_node_update<6><<<nu_blocks, 256, 0, stream>>>(x0, rt1, bs1, sb, deg, xA, xAb);

  // layer 2 (ci=96)
  (void)hipMemsetAsync(sb, 0, (size_t)NODES * 96 * 4, stream);
  k_edge_gemm<96, 9216><<<gg, TPB, 0, stream>>>(eatt, w2a, b2a, xAb, src, dst, wbp2, b2b, sb);
  k_node_update<96><<<nu_blocks, 256, 0, stream>>>(xA, rt2, bs2, sb, deg, xB, xBb);

  // layer 3 (ci=96)
  (void)hipMemsetAsync(sb, 0, (size_t)NODES * 96 * 4, stream);
  k_edge_gemm<96, 9216><<<gg, TPB, 0, stream>>>(eatt, w3a, b3a, xBb, src, dst, wbp3, b3b, sb);
  k_node_update<96><<<nu_blocks, 256, 0, stream>>>(xB, rt3, bs3, sb, deg, xA, xAb);

  // global mean pool + readout MLP
  (void)hipMemsetAsync(psum, 0, (size_t)GRAPHS * 96 * 4, stream);
  (void)hipMemsetAsync(cnt,  0, (size_t)GRAPHS * 4, stream);
  k_pool<<<nu_blocks, 256, 0, stream>>>(xA, batch, psum, cnt);
  k_readout<<<1, GRAPHS * 96, 0, stream>>>(psum, cnt, wm1, bm1, wm2, bm2, (float*)d_out);
}
